// TemporalHGTConv_57552561766833
// MI455X (gfx1250) — matrix operand order repaired and checked by reading, not compiled, not run on hardware
//
#include <hip/hip_runtime.h>
#include <hip/hip_bf16.h>

#define D_   256
#define H_   8
#define DH_  32
#define NOP_ 20000
#define NM_  500
#define NE_  200000

typedef __attribute__((ext_vector_type(16))) _Float16 v16h;
typedef __attribute__((ext_vector_type(8)))  float    v8f;
typedef __attribute__((ext_vector_type(4)))  unsigned int u32x4;
typedef __attribute__((ext_vector_type(4)))  float    f32x4;

union V16H { v16h v; _Float16 h[16]; u32x4 q[2]; };

__device__ inline v8f wmma_f16(v16h a, v16h b, v8f c) {
  // D = A(16x32 f16) x B(32x16 f16) + C(16x16 f32)
  return __builtin_amdgcn_wmma_f32_16x16x32_f16(false, a, false, b, (short)0, c, false, false);
}

// order-preserving float <-> uint encoding for atomicMax-based segment max
__device__ inline unsigned fenc(float f) {
  unsigned u = __float_as_uint(f);
  return (u & 0x80000000u) ? ~u : (u | 0x80000000u);
}
__device__ inline float fdec(unsigned k) {
  return (k & 0x80000000u) ? __uint_as_float(k & 0x7fffffffu) : __uint_as_float(~k);
}

__device__ inline void cvt4(_Float16* dst, f32x4 f) {
  dst[0] = (_Float16)f.x; dst[1] = (_Float16)f.y;
  dst[2] = (_Float16)f.z; dst[3] = (_Float16)f.w;
}

// ---------------- elementwise helpers ----------------
__global__ void k_copy_f32(const float* __restrict__ in, float* __restrict__ out, int n) {
  int i = blockIdx.x * blockDim.x + threadIdx.x;
  if (i < n) out[i] = in[i];
}
__global__ void k_zero_f32(float* __restrict__ out, int n) {
  int i = blockIdx.x * blockDim.x + threadIdx.x;
  if (i < n) out[i] = 0.0f;
}
__global__ void k_cvt_f16(const float* __restrict__ in, _Float16* __restrict__ out, int n) {
  int i = blockIdx.x * blockDim.x + threadIdx.x;
  if (i < n) out[i] = (_Float16)in[i];
}
__global__ void k_gelu_f16(const float* __restrict__ in, _Float16* __restrict__ out, int n) {
  int i = blockIdx.x * blockDim.x + threadIdx.x;
  if (i < n) {
    float x = in[i];
    out[i] = (_Float16)(0.5f * x * (1.0f + erff(x * 0.70710678118f)));
  }
}
__global__ void k_init_softmax(unsigned* __restrict__ amax, float* __restrict__ denom, int n) {
  int i = blockIdx.x * blockDim.x + threadIdx.x;
  if (i < n) { amax[i] = 0x007FFFFFu; denom[i] = 0.0f; }  // enc(-inf)
}

// ---------------- temporal encoding + scatter ----------------
// one block (256 threads) per edge; pt = temb @ Wt.T + bt, LayerNorm, atomic scatter-add at src
__global__ void k_temporal(const float* __restrict__ dtv, const int* __restrict__ ei,
                           const float* __restrict__ Wt, const float* __restrict__ bt,
                           const float* __restrict__ ln_g, const float* __restrict__ ln_b,
                           float* __restrict__ xop2) {
  __shared__ float red[256];
  int e = blockIdx.x;
  int j = threadIdx.x;
  float dt = dtv[e];
  const float fr[8] = {1.0f, 0.42169650342f, 0.17782794100f, 0.07498942093f,
                       0.03162277660f, 0.01333521432f, 0.00562341325f, 0.00237137371f};
  float temb[16];
#pragma unroll
  for (int i = 0; i < 8; ++i) {
    float s, c;
    __sincosf(dt * fr[i], &s, &c);
    temb[2 * i] = s; temb[2 * i + 1] = c;
  }
  const float* wr = Wt + j * 16;
  float pt = bt[j];
#pragma unroll
  for (int c = 0; c < 16; ++c) pt += temb[c] * wr[c];
  red[j] = pt; __syncthreads();
  for (int s = 128; s > 0; s >>= 1) { if (j < s) red[j] += red[j + s]; __syncthreads(); }
  float mu = red[0] * (1.0f / 256.0f);
  __syncthreads();
  float dv = pt - mu;
  red[j] = dv * dv; __syncthreads();
  for (int s = 128; s > 0; s >>= 1) { if (j < s) red[j] += red[j + s]; __syncthreads(); }
  float var = red[0] * (1.0f / 256.0f);
  float val = dv * rsqrtf(var + 1e-5f) * ln_g[j] + ln_b[j];
  atomicAdd(&xop2[(size_t)ei[e] * 256 + j], val);
}

// ---------------- generic 256-K WMMA GEMM: C[M,256] = X @ W^T + bias ----------------
// block = 256 threads = 8 waves; wave owns one 16x16 tile; grid = (ceil(M/16), 2)
__global__ void k_gemm256(const _Float16* __restrict__ X, const _Float16* __restrict__ W,
                          const float* __restrict__ bias, float* __restrict__ C, int M) {
  int wave = threadIdx.x >> 5, lane = threadIdx.x & 31;
  int half = lane >> 4, l16 = lane & 15;
  int tileM = blockIdx.x;
  int tileN = blockIdx.y * 8 + wave;
  int rowA = tileM * 16 + l16; if (rowA >= M) rowA = M - 1;
  int rowB = tileN * 16 + l16;
  const _Float16* Ap = X + (size_t)rowA * 256 + half * 8;
  const _Float16* Bp = W + (size_t)rowB * 256 + half * 8;
  v8f acc = {};
#pragma unroll
  for (int kb = 0; kb < 256; kb += 32) {
    V16H a, b;
    a.q[0] = *(const u32x4*)(Ap + kb);
    a.q[1] = *(const u32x4*)(Ap + kb + 16);
    b.q[0] = *(const u32x4*)(Bp + kb);
    b.q[1] = *(const u32x4*)(Bp + kb + 16);
    acc = wmma_f16(a.v, b.v, acc);
  }
  int col = tileN * 16 + l16;
  float bc = bias[col];
#pragma unroll
  for (int r = 0; r < 8; ++r) {
    int row = tileM * 16 + r + half * 8;
    if (row < M) C[(size_t)row * 256 + col] = acc[r] + bc;
  }
}

// ---------------- final GEMM + exact-erf GELU input (pre-converted) + gated skip ----------------
__global__ void k_gemm_final(const _Float16* __restrict__ G, const _Float16* __restrict__ W,
                             const float* __restrict__ bias, const float* __restrict__ Xres,
                             const float* __restrict__ skipv, int sidx,
                             float* __restrict__ OUT, int M) {
  int wave = threadIdx.x >> 5, lane = threadIdx.x & 31;
  int half = lane >> 4, l16 = lane & 15;
  int tileM = blockIdx.x;
  int tileN = blockIdx.y * 8 + wave;
  int rowA = tileM * 16 + l16; if (rowA >= M) rowA = M - 1;
  int rowB = tileN * 16 + l16;
  const _Float16* Ap = G + (size_t)rowA * 256 + half * 8;
  const _Float16* Bp = W + (size_t)rowB * 256 + half * 8;
  v8f acc = {};
#pragma unroll
  for (int kb = 0; kb < 256; kb += 32) {
    V16H a, b;
    a.q[0] = *(const u32x4*)(Ap + kb);
    a.q[1] = *(const u32x4*)(Ap + kb + 16);
    b.q[0] = *(const u32x4*)(Bp + kb);
    b.q[1] = *(const u32x4*)(Bp + kb + 16);
    acc = wmma_f16(a.v, b.v, acc);
  }
  int col = tileN * 16 + l16;
  float bc = bias[col];
  float ag = 1.0f / (1.0f + __expf(-skipv[sidx]));
#pragma unroll
  for (int r = 0; r < 8; ++r) {
    int row = tileM * 16 + r + half * 8;
    if (row < M) {
      float o = acc[r] + bc;
      OUT[(size_t)row * 256 + col] = ag * o + (1.0f - ag) * Xres[(size_t)row * 256 + col];
    }
  }
}

// ---------------- pass A: per-edge-type attention logits + segment max ----------------
// block = 256 threads = 8 waves, wave w handles head w for groups of 16 edges
__global__ void k_edge_alpha(const int* __restrict__ ei, const float* __restrict__ Ksrc,
                             const float* __restrict__ Qdst, const _Float16* __restrict__ arel,
                             const float* __restrict__ prel, float* __restrict__ alphaBuf,
                             unsigned* __restrict__ amax, int numGroups) {
  int h = threadIdx.x >> 5, lane = threadIdx.x & 31;
  int half = lane >> 4, l16 = lane & 15;
  // B fragments for a_rel[h] (32x32), two 16-wide N tiles, held in registers
  V16H b0, b1;
  const _Float16* ab = arel + h * DH_ * DH_;
#pragma unroll
  for (int j = 0; j < 16; ++j) {
    int d = half * 8 + j + (j >= 8 ? 8 : 0);
    b0.h[j] = ab[d * 32 + l16];
    b1.h[j] = ab[d * 32 + 16 + l16];
  }
  float pr = prel[h] * 0.17677669529f;  // p_rel * 1/sqrt(DH)
  for (int g = blockIdx.x; g < numGroups; g += gridDim.x) {
    int eb = g * 16;
    int srow = ei[eb + l16];  // A row M = lane&15
    const float* kp = Ksrc + (size_t)srow * 256 + h * 32 + half * 8;
    V16H a;
    cvt4(a.h + 0,  *(const f32x4*)(kp));
    cvt4(a.h + 4,  *(const f32x4*)(kp + 4));
    cvt4(a.h + 8,  *(const f32x4*)(kp + 16));
    cvt4(a.h + 12, *(const f32x4*)(kp + 20));
    v8f c0 = {}, c1 = {};
    c0 = wmma_f16(a.v, b0.v, c0);   // k_e[:, 0:16]
    c1 = wmma_f16(a.v, b1.v, c1);   // k_e[:,16:32]
    // alpha[m,h] = sum_f k_e[m,f]*q[dst[m],h,f]; lane covers f=l16 and f=16+l16
    float part[8];
#pragma unroll
    for (int r = 0; r < 8; ++r) {
      int drow = ei[NE_ + eb + r + half * 8];
      const float* qp = Qdst + (size_t)drow * 256 + h * 32;
      part[r] = c0[r] * qp[l16] + c1[r] * qp[16 + l16];
    }
#pragma unroll
    for (int r = 0; r < 8; ++r) {   // butterfly sum over the 16 lanes of this half
      float p = part[r];
      p += __shfl_xor(p, 1, 32);
      p += __shfl_xor(p, 2, 32);
      p += __shfl_xor(p, 4, 32);
      p += __shfl_xor(p, 8, 32);
      part[r] = p;
    }
    float aval = part[0];
#pragma unroll
    for (int r = 1; r < 8; ++r) if (l16 == r) aval = part[r];
    if (l16 < 8) {
      int m = eb + l16 + half * 8;
      float alpha = aval * pr;
      alphaBuf[(size_t)m * 8 + h] = alpha;
      atomicMax(&amax[(size_t)ei[NE_ + m] * 8 + h], fenc(alpha));
    }
  }
}

// ---------------- pass B: exp(alpha - max) and segment sum (overwrites alphaBuf with expa) ----------------
__global__ void k_edge_denom(const int* __restrict__ ei, float* __restrict__ alphaBuf,
                             const unsigned* __restrict__ amax, float* __restrict__ denom) {
  int t = blockIdx.x * blockDim.x + threadIdx.x;
  if (t >= NE_ * H_) return;
  int m = t >> 3, h = t & 7;
  int d = ei[NE_ + m];
  float ex = __expf(alphaBuf[t] - fdec(amax[(size_t)d * 8 + h]));
  alphaBuf[t] = ex;
  atomicAdd(&denom[(size_t)d * 8 + h], ex);
}

// ---------------- pass C: v-transform via WMMA, weighted scatter into out[dst] ----------------
__global__ void k_edge_out(const int* __restrict__ ei, const float* __restrict__ Vsrc,
                           const _Float16* __restrict__ mrel, const float* __restrict__ expa,
                           const float* __restrict__ denom, float* __restrict__ outDst,
                           int numGroups) {
  int h = threadIdx.x >> 5, lane = threadIdx.x & 31;
  int half = lane >> 4, l16 = lane & 15;
  V16H b0, b1;
  const _Float16* mb = mrel + h * DH_ * DH_;
#pragma unroll
  for (int j = 0; j < 16; ++j) {
    int d = half * 8 + j + (j >= 8 ? 8 : 0);
    b0.h[j] = mb[d * 32 + l16];
    b1.h[j] = mb[d * 32 + 16 + l16];
  }
  for (int g = blockIdx.x; g < numGroups; g += gridDim.x) {
    int eb = g * 16;
    int srow = ei[eb + l16];
    const float* vp = Vsrc + (size_t)srow * 256 + h * 32 + half * 8;
    V16H a;
    cvt4(a.h + 0,  *(const f32x4*)(vp));
    cvt4(a.h + 4,  *(const f32x4*)(vp + 4));
    cvt4(a.h + 8,  *(const f32x4*)(vp + 16));
    cvt4(a.h + 12, *(const f32x4*)(vp + 20));
    v8f c0 = {}, c1 = {};
    c0 = wmma_f16(a.v, b0.v, c0);
    c1 = wmma_f16(a.v, b1.v, c1);
#pragma unroll
    for (int r = 0; r < 8; ++r) {
      int m = eb + r + half * 8;
      int drow = ei[NE_ + m];
      float w = expa[(size_t)m * 8 + h] / denom[(size_t)drow * 8 + h];
      atomicAdd(&outDst[(size_t)drow * 256 + h * 32 + l16],      c0[r] * w);
      atomicAdd(&outDst[(size_t)drow * 256 + h * 32 + 16 + l16], c1[r] * w);
    }
  }
}

// ---------------- host orchestration ----------------
extern "C" void kernel_launch(void* const* d_in, const int* in_sizes, int n_in,
                              void* d_out, int out_size, void* d_ws, size_t ws_size,
                              hipStream_t stream) {
  (void)in_sizes; (void)n_in; (void)out_size; (void)ws_size;
  const float* x_op      = (const float*)d_in[0];
  const float* x_machine = (const float*)d_in[1];
  const float* delta_t   = (const float*)d_in[2];
  const int* eis[4] = {(const int*)d_in[3], (const int*)d_in[4],
                       (const int*)d_in[5], (const int*)d_in[6]};
  const float* Wt = (const float*)d_in[7];
  const float* bt = (const float*)d_in[8];
  const float* ln_g = (const float*)d_in[9];
  const float* ln_b = (const float*)d_in[10];
  const float* Wk = (const float*)d_in[11]; const float* bk = (const float*)d_in[12];
  const float* Wq = (const float*)d_in[13]; const float* bq = (const float*)d_in[14];
  const float* Wv = (const float*)d_in[15]; const float* bv = (const float*)d_in[16];
  const float* Wa = (const float*)d_in[17]; const float* ba = (const float*)d_in[18];
  const float* skip  = (const float*)d_in[19];
  const float* a_rel = (const float*)d_in[20];
  const float* m_rel = (const float*)d_in[21];
  const float* p_rel = (const float*)d_in[22];

  char* wp = (char*)d_ws;
  auto carve = [&](size_t bytes) -> void* {
    void* p = (void*)wp;
    wp += (bytes + 255) & ~(size_t)255;
    return p;
  };
  float* xop2  = (float*)carve((size_t)NOP_ * 256 * 4);
  float* outOp = (float*)carve((size_t)NOP_ * 256 * 4);
  float* outM  = (float*)carve((size_t)NM_  * 256 * 4);
  float* Kop = (float*)carve((size_t)NOP_ * 256 * 4);
  float* Qop = (float*)carve((size_t)NOP_ * 256 * 4);
  float* Vop = (float*)carve((size_t)NOP_ * 256 * 4);
  float* Km  = (float*)carve((size_t)NM_ * 256 * 4);
  float* Qm  = (float*)carve((size_t)NM_ * 256 * 4);
  float* Vm  = (float*)carve((size_t)NM_ * 256 * 4);
  float*    alphaB = (float*)carve((size_t)NE_ * 8 * 4);
  unsigned* amax   = (unsigned*)carve((size_t)NOP_ * 8 * 4);
  float*    denom  = (float*)carve((size_t)NOP_ * 8 * 4);
  _Float16* xop16 = (_Float16*)carve((size_t)NOP_ * 256 * 2);
  _Float16* xm16  = (_Float16*)carve((size_t)NM_  * 256 * 2);
  _Float16* Wk16 = (_Float16*)carve((size_t)2 * 256 * 256 * 2);
  _Float16* Wq16 = (_Float16*)carve((size_t)2 * 256 * 256 * 2);
  _Float16* Wv16 = (_Float16*)carve((size_t)2 * 256 * 256 * 2);
  _Float16* Wa16 = (_Float16*)carve((size_t)2 * 256 * 256 * 2);
  _Float16* arel16 = (_Float16*)carve((size_t)4 * 8 * 32 * 32 * 2);
  _Float16* mrel16 = (_Float16*)carve((size_t)4 * 8 * 32 * 32 * 2);
  _Float16* gop16 = (_Float16*)carve((size_t)NOP_ * 256 * 2);
  _Float16* gm16  = (_Float16*)carve((size_t)NM_  * 256 * 2);

  auto cdiv = [](int a, int b) { return (a + b - 1) / b; };
  const int nOpF = NOP_ * 256, nMF = NM_ * 256;

  // 1. temporal encoding + scatter onto a private copy of x_op
  k_copy_f32<<<cdiv(nOpF, 256), 256, 0, stream>>>(x_op, xop2, nOpF);
  k_zero_f32<<<cdiv(nOpF, 256), 256, 0, stream>>>(outOp, nOpF);
  k_zero_f32<<<cdiv(nMF, 256), 256, 0, stream>>>(outM, nMF);
  k_temporal<<<NE_, 256, 0, stream>>>(delta_t, eis[0], Wt, bt, ln_g, ln_b, xop2);

  // 2. f16 staging for WMMA
  k_cvt_f16<<<cdiv(nOpF, 256), 256, 0, stream>>>(xop2, xop16, nOpF);
  k_cvt_f16<<<cdiv(nMF, 256), 256, 0, stream>>>(x_machine, xm16, nMF);
  const int nW = 2 * 256 * 256;
  k_cvt_f16<<<cdiv(nW, 256), 256, 0, stream>>>(Wk, Wk16, nW);
  k_cvt_f16<<<cdiv(nW, 256), 256, 0, stream>>>(Wq, Wq16, nW);
  k_cvt_f16<<<cdiv(nW, 256), 256, 0, stream>>>(Wv, Wv16, nW);
  k_cvt_f16<<<cdiv(nW, 256), 256, 0, stream>>>(Wa, Wa16, nW);
  const int nR = 4 * 8 * 32 * 32;
  k_cvt_f16<<<cdiv(nR, 256), 256, 0, stream>>>(a_rel, arel16, nR);
  k_cvt_f16<<<cdiv(nR, 256), 256, 0, stream>>>(m_rel, mrel16, nR);

  // 3. QKV projections (WMMA)
  dim3 gOp(cdiv(NOP_, 16), 2), gM(cdiv(NM_, 16), 2);
  k_gemm256<<<gOp, 256, 0, stream>>>(xop16, Wk16,         bk,       Kop, NOP_);
  k_gemm256<<<gOp, 256, 0, stream>>>(xop16, Wq16,         bq,       Qop, NOP_);
  k_gemm256<<<gOp, 256, 0, stream>>>(xop16, Wv16,         bv,       Vop, NOP_);
  k_gemm256<<<gM,  256, 0, stream>>>(xm16,  Wk16 + 65536, bk + 256, Km,  NM_);
  k_gemm256<<<gM,  256, 0, stream>>>(xm16,  Wq16 + 65536, bq + 256, Qm,  NM_);
  k_gemm256<<<gM,  256, 0, stream>>>(xm16,  Wv16 + 65536, bv + 256, Vm,  NM_);

  // 4. HGT per-edge-type attention: softmax is per edge type; out accumulates
  const int smap[4] = {0, 0, 0, 1}, dmap[4] = {0, 0, 1, 0};
  const float* Kt[2] = {Kop, Km};
  const float* Qt[2] = {Qop, Qm};
  const float* Vt[2] = {Vop, Vm};
  float* Ot[2] = {outOp, outM};
  const int Ns[2] = {NOP_, NM_};
  const int numGroups = NE_ / 16;
  for (int e = 0; e < 4; ++e) {
    int s = smap[e], d = dmap[e];
    int n = Ns[d] * 8;
    k_init_softmax<<<cdiv(n, 256), 256, 0, stream>>>(amax, denom, n);
    k_edge_alpha<<<2048, 256, 0, stream>>>(eis[e], Kt[s], Qt[d],
                                           arel16 + e * 8 * 32 * 32, p_rel + e * 8,
                                           alphaB, amax, numGroups);
    k_edge_denom<<<cdiv(NE_ * 8, 256), 256, 0, stream>>>(eis[e], alphaB, amax, denom);
    k_edge_out<<<2048, 256, 0, stream>>>(eis[e], Vt[s], mrel16 + e * 8 * 32 * 32,
                                         alphaB, denom, Ot[d], numGroups);
  }

  // 5. exact GELU -> f16, output projection + gated skip (WMMA)
  k_gelu_f16<<<cdiv(nOpF, 256), 256, 0, stream>>>(outOp, gop16, nOpF);
  k_gelu_f16<<<cdiv(nMF, 256), 256, 0, stream>>>(outM, gm16, nMF);
  float* res_op = (float*)d_out;
  float* res_m  = res_op + (size_t)NOP_ * 256;
  k_gemm_final<<<gOp, 256, 0, stream>>>(gop16, Wa16,         ba,       xop2,      skip, 0, res_op, NOP_);
  k_gemm_final<<<gM,  256, 0, stream>>>(gm16,  Wa16 + 65536, ba + 256, x_machine, skip, 1, res_m,  NM_);
}
